// HyperGCN_Net_81106162418145
// MI455X (gfx1250) — matrix operand order repaired and checked by reading, compiled, not run on hardware
//
#include <hip/hip_runtime.h>

#define N_NODES 100000
#define N_EDGES 20000
#define NNZ_CNT 3200000LL
#define F_IN    512
#define HID     64
#define CLS     70
#define CLS_PAD 80

typedef __bf16 v16bf __attribute__((ext_vector_type(16)));
typedef float  v8f   __attribute__((ext_vector_type(8)));

// ---------------------------------------------------------------- utilities
__global__ void zero_f32(float* __restrict__ p, long long n) {
    long long i = blockIdx.x * (long long)blockDim.x + threadIdx.x;
    long long stride = (long long)gridDim.x * blockDim.x;
    for (; i < n; i += stride) p[i] = 0.0f;
}

__global__ void degrees_kernel(const long long* __restrict__ src,
                               const long long* __restrict__ edg,
                               float* __restrict__ deg_n,
                               float* __restrict__ deg_e,
                               long long nnz) {
    long long i = blockIdx.x * (long long)blockDim.x + threadIdx.x;
    if (i >= nnz) return;
    atomicAdd(&deg_n[(int)src[i]], 1.0f);
    atomicAdd(&deg_e[(int)edg[i]], 1.0f);
}

__global__ void invert_kernel(float* __restrict__ d, int n) {
    int i = blockIdx.x * blockDim.x + threadIdx.x;
    if (i >= n) return;
    float v = d[i];
    d[i] = (v > 0.0f) ? (1.0f / v) : 0.0f;
}

// Pack W[K,N] (fp32, row-major) into per-lane bf16 B-fragments.
// Fragment layout (16-bit B 32x16): lane<16 -> column=lane, elems 0..7 = K kstep*32+0..7, 8..15 = K kstep*32+16..23
//                                   lane>=16 -> column=lane-16, elems 0..7 = K kstep*32+8..15, 8..15 = K kstep*32+24..31
// Wp index = ((kstep*NT + t)*32 + lane)*16 + j ; columns >= N zero-padded.
__global__ void pack_w_kernel(const float* __restrict__ W, __bf16* __restrict__ Wp,
                              int K, int N, int NT) {
    int gid = blockIdx.x * blockDim.x + threadIdx.x;
    int ksteps = K >> 5;
    int total = ksteps * NT * 32 * 16;
    if (gid >= total) return;
    int j     = gid & 15;
    int rest  = gid >> 4;
    int lane  = rest & 31;
    int rest2 = rest >> 5;
    int t     = rest2 % NT;
    int kstep = rest2 / NT;
    int k = kstep * 32 + ((lane >> 4) << 3) + (j & 7) + ((j >> 3) << 4);
    int n = t * 16 + (lane & 15);
    float v = (n < N) ? W[(size_t)k * N + n] : 0.0f;
    Wp[gid] = (__bf16)v;
}

// ---------------------------------------------------------------- WMMA GEMM
// C[M x (NT*16)] = A[M x K] * W  (W pre-packed bf16 fragments). ldc = NT*16 or given.
// 8 waves / 256-thread block, each wave computes a 16-row strip, NT 16x16 tiles wide.
template<int NT>
__global__ void wmma_gemm_kernel(const float* __restrict__ A,
                                 const __bf16* __restrict__ Wp,
                                 float* __restrict__ C,
                                 int M, int K, int ldc) {
    const int wave = threadIdx.x >> 5;
    const int lane = threadIdx.x & 31;
    const int row_base = (blockIdx.x * (blockDim.x >> 5) + wave) * 16;
    if (row_base >= M) return;   // wave-uniform: EXEC stays all-ones for WMMA

    const int mrow  = row_base + (lane & 15);
    const int khalf = (lane >> 4) << 3;        // 0 or 8
    const float* __restrict__ arow = A + (size_t)mrow * K;
    const v16bf* __restrict__ wp = (const v16bf*)Wp;

    v8f zero = {};
    v8f acc[NT];
#pragma unroll
    for (int t = 0; t < NT; ++t) acc[t] = zero;

    const int ksteps = K >> 5;
    for (int ks = 0; ks < ksteps; ++ks) {
        const int k0 = ks << 5;
        // A fragment: 16-bit A 16x32 layout.
        // lane<16: elems 0..7 = K k0+0..7,  elems 8..15 = K k0+16..23
        // lane>=16: elems 0..7 = K k0+8..15, elems 8..15 = K k0+24..31
        const float4 f0 = *(const float4*)(arow + k0 + khalf);
        const float4 f1 = *(const float4*)(arow + k0 + khalf + 4);
        const float4 f2 = *(const float4*)(arow + k0 + 16 + khalf);
        const float4 f3 = *(const float4*)(arow + k0 + 16 + khalf + 4);
        v16bf a;
        a[0]  = (__bf16)f0.x; a[1]  = (__bf16)f0.y; a[2]  = (__bf16)f0.z; a[3]  = (__bf16)f0.w;
        a[4]  = (__bf16)f1.x; a[5]  = (__bf16)f1.y; a[6]  = (__bf16)f1.z; a[7]  = (__bf16)f1.w;
        a[8]  = (__bf16)f2.x; a[9]  = (__bf16)f2.y; a[10] = (__bf16)f2.z; a[11] = (__bf16)f2.w;
        a[12] = (__bf16)f3.x; a[13] = (__bf16)f3.y; a[14] = (__bf16)f3.z; a[15] = (__bf16)f3.w;
#pragma unroll
        for (int t = 0; t < NT; ++t) {
            v16bf b = wp[(size_t)(ks * NT + t) * 32 + lane];
            acc[t] = __builtin_amdgcn_wmma_f32_16x16x32_bf16(
                false, a, false, b, (short)0, acc[t], false, false);
        }
    }

    // C layout: lane<16 -> VGPR r = (M=row_base+r, N=lane); lane>=16 -> (M=row_base+8+r, N=lane-16)
    const int m_off = (lane >> 4) << 3;
    const int n_col = lane & 15;
#pragma unroll
    for (int t = 0; t < NT; ++t) {
        float* crow = C + (size_t)(row_base + m_off) * ldc + t * 16 + n_col;
#pragma unroll
        for (int r = 0; r < 8; ++r) crow[(size_t)r * ldc] = acc[t][r];
    }
}

// ---------------------------------------------------------------- scatters
// he[e, f] += h[s, f]   (node -> hyperedge), 2 features per thread
__global__ void scatter_n2e_kernel(const long long* __restrict__ src,
                                   const long long* __restrict__ edg,
                                   const float* __restrict__ h, int ldh,
                                   float* __restrict__ he, int lde,
                                   int F, int cpr, long long total) {
    long long gid = blockIdx.x * (long long)blockDim.x + threadIdx.x;
    if (gid >= total) return;
    long long i = gid / cpr;
    int f = (int)(gid % cpr) * 2;
    int s = (int)src[i], e = (int)edg[i];
    const float* hp = h + (size_t)s * ldh + f;
    float* hep = he + (size_t)e * lde + f;
    atomicAdd(hep, hp[0]);
    if (f + 1 < F) atomicAdd(hep + 1, hp[1]);
}

// out[s, f] += Binv[e] * he[e, f]   (hyperedge -> node, B^{-1} folded in)
__global__ void scatter_e2n_kernel(const long long* __restrict__ src,
                                   const long long* __restrict__ edg,
                                   const float* __restrict__ he, int lde,
                                   const float* __restrict__ binv,
                                   float* __restrict__ out, int ldo,
                                   int F, int cpr, long long total) {
    long long gid = blockIdx.x * (long long)blockDim.x + threadIdx.x;
    if (gid >= total) return;
    long long i = gid / cpr;
    int f = (int)(gid % cpr) * 2;
    int s = (int)src[i], e = (int)edg[i];
    float bi = binv[e];
    const float* hep = he + (size_t)e * lde + f;
    float* op = out + (size_t)s * ldo + f;
    atomicAdd(op, bi * hep[0]);
    if (f + 1 < F) atomicAdd(op + 1, bi * hep[1]);
}

// data[r, f] = act(Dinv[r] * data[r, f] + bias[f])
__global__ void finalize_kernel(float* __restrict__ data,
                                const float* __restrict__ dinv,
                                const float* __restrict__ bias,
                                long long rows, int F, int ld, int do_relu) {
    long long gid = blockIdx.x * (long long)blockDim.x + threadIdx.x;
    if (gid >= rows * F) return;
    long long r = gid / F;
    int f = (int)(gid % F);
    float v = dinv[r] * data[(size_t)r * ld + f] + bias[f];
    if (do_relu) v = fmaxf(v, 0.0f);
    data[(size_t)r * ld + f] = v;
}

// ---------------------------------------------------------------- launcher
extern "C" void kernel_launch(void* const* d_in, const int* in_sizes, int n_in,
                              void* d_out, int out_size, void* d_ws, size_t ws_size,
                              hipStream_t stream) {
    const float* x        = (const float*)d_in[0];
    const long long* hidx = (const long long*)d_in[1];
    const float* W1       = (const float*)d_in[2];
    const float* b1       = (const float*)d_in[3];
    const float* W2       = (const float*)d_in[4];
    const float* b2       = (const float*)d_in[5];
    float* out            = (float*)d_out;

    const long long* src = hidx;            // row 0: node ids
    const long long* edg = hidx + NNZ_CNT;  // row 1: hyperedge ids

    // ---- workspace carving (256B-aligned) ----
    char* w = (char*)d_ws;
    size_t off = 0;
    auto carve = [&](size_t bytes) -> void* {
        void* p = w + off;
        off += (bytes + 255) & ~(size_t)255;
        return p;
    };
    float*  deg_e = (float*)carve((size_t)N_EDGES * 4);
    float*  deg_n = (float*)carve((size_t)N_NODES * 4);
    float*  hbig  = (float*)carve((size_t)N_NODES * CLS_PAD * 4); // h1 (ld=64) then h2 (ld=80)
    float*  hmid  = (float*)carve((size_t)N_NODES * HID * 4);
    float*  he1   = (float*)carve((size_t)N_EDGES * HID * 4);
    float*  he2   = (float*)carve((size_t)N_EDGES * CLS * 4);
    __bf16* wp1   = (__bf16*)carve((size_t)(F_IN / 32) * 4 * 32 * 16 * 2);
    __bf16* wp2   = (__bf16*)carve((size_t)(HID  / 32) * 5 * 32 * 16 * 2);

    const int B = 256;

    // ---- zero accumulation buffers (every call: graph-replay safe) ----
    zero_f32<<<2048, B, 0, stream>>>(deg_e, (long long)N_EDGES);
    zero_f32<<<2048, B, 0, stream>>>(deg_n, (long long)N_NODES);
    zero_f32<<<2048, B, 0, stream>>>(he1,  (long long)N_EDGES * HID);
    zero_f32<<<2048, B, 0, stream>>>(hmid, (long long)N_NODES * HID);
    zero_f32<<<2048, B, 0, stream>>>(he2,  (long long)N_EDGES * CLS);
    zero_f32<<<2048, B, 0, stream>>>(out,  (long long)N_NODES * CLS);

    // ---- degrees -> inverses ----
    degrees_kernel<<<(int)((NNZ_CNT + B - 1) / B), B, 0, stream>>>(src, edg, deg_n, deg_e, NNZ_CNT);
    invert_kernel<<<(N_EDGES + B - 1) / B, B, 0, stream>>>(deg_e, N_EDGES);
    invert_kernel<<<(N_NODES + B - 1) / B, B, 0, stream>>>(deg_n, N_NODES);

    // ---- pack weights into bf16 WMMA B-fragments ----
    {
        int tot1 = (F_IN / 32) * 4 * 32 * 16;
        pack_w_kernel<<<(tot1 + B - 1) / B, B, 0, stream>>>(W1, wp1, F_IN, HID, 4);
        int tot2 = (HID / 32) * 5 * 32 * 16;
        pack_w_kernel<<<(tot2 + B - 1) / B, B, 0, stream>>>(W2, wp2, HID, CLS, 5);
    }

    const int gemm_grid = (N_NODES + 127) / 128; // 8 waves x 16 rows per block

    // ---- conv1: h1 = x @ W1 (bf16 WMMA) ----
    wmma_gemm_kernel<4><<<gemm_grid, B, 0, stream>>>(x, wp1, hbig, N_NODES, F_IN, HID);
    {   // node -> edge, edge -> node (Binv folded), finalize (+b1, ReLU, Dinv)
        long long tot = NNZ_CNT * (HID / 2);
        int grid = (int)((tot + B - 1) / B);
        scatter_n2e_kernel<<<grid, B, 0, stream>>>(src, edg, hbig, HID, he1, HID, HID, HID / 2, tot);
        scatter_e2n_kernel<<<grid, B, 0, stream>>>(src, edg, he1, HID, deg_e, hmid, HID, HID, HID / 2, tot);
        long long fin = (long long)N_NODES * HID;
        finalize_kernel<<<(int)((fin + B - 1) / B), B, 0, stream>>>(hmid, deg_n, b1,
                                                                    (long long)N_NODES, HID, HID, 1);
    }

    // ---- conv2: h2 = hmid @ W2 (bf16 WMMA, N padded 70 -> 80) ----
    wmma_gemm_kernel<5><<<gemm_grid, B, 0, stream>>>(hmid, wp2, hbig, N_NODES, HID, CLS_PAD);
    {
        int cpr = (CLS + 1) / 2; // 35
        long long tot = NNZ_CNT * cpr;
        int grid = (int)((tot + B - 1) / B);
        scatter_n2e_kernel<<<grid, B, 0, stream>>>(src, edg, hbig, CLS_PAD, he2, CLS, CLS, cpr, tot);
        scatter_e2n_kernel<<<grid, B, 0, stream>>>(src, edg, he2, CLS, deg_e, out, CLS, CLS, cpr, tot);
        long long fin = (long long)N_NODES * CLS;
        finalize_kernel<<<(int)((fin + B - 1) / B), B, 0, stream>>>(out, deg_n, b2,
                                                                    (long long)N_NODES, CLS, CLS, 0);
    }
}
